// RPNHead_3882650435978
// MI455X (gfx1250) — compile-verified, matrix operand order checked by reading, not run on hardware
//
#include <hip/hip_runtime.h>

// Problem constants
#define B_    32
#define CIN   1024
#define H_    37
#define W_    37
#define HW    (H_ * W_)        // 1369
#define DIM   512
#define COUT  120
#define K1    (CIN * 9)        // 9216 : implicit-GEMM K of the 3x3 conv
#define NTOT  (B_ * HW)        // 43808 spatial positions

// Tiling
#define BM  128
#define BN  128
#define BK  32
#define LDK 40                 // padded LDS row stride in halves (80B rows, 16B-aligned chunks)
#define NSTEPS (9 * (CIN / BK))   // 288 pipeline steps

typedef __bf16 v16bf __attribute__((ext_vector_type(16)));
typedef __bf16 v8bf  __attribute__((ext_vector_type(8)));
typedef float  v8f   __attribute__((ext_vector_type(8)));

union Frag  { v16bf v; v8bf h[2]; };
union Pack8 { __bf16 b[8]; uint4 u; };

__device__ __forceinline__ unsigned short bf16_bits(float f) {
  union { float f; unsigned u; } a; a.f = f;
  unsigned r = a.u + 0x7FFFu + ((a.u >> 16) & 1u);   // round-to-nearest-even
  return (unsigned short)(r >> 16);
}
__device__ __forceinline__ __bf16 f2bf(float f) {
  union { unsigned short s; __bf16 b; } o; o.s = bf16_bits(f);
  return o.b;
}
__device__ __forceinline__ unsigned pack2(float a, float b) {
  return (unsigned)bf16_bits(a) | ((unsigned)bf16_bits(b) << 16);
}

// =========================================================================
// Stage 0a: fmap fp32 NCHW -> bf16 NCHW (one-shot, fully vectorized)
// =========================================================================
__global__ __launch_bounds__(256)
void prep_fmap_bf16(const float* __restrict__ src, unsigned short* __restrict__ dst) {
  size_t i8 = ((size_t)blockIdx.x * 256 + threadIdx.x) * 8;
  float4 f0 = *(const float4*)(src + i8);
  float4 f1 = *(const float4*)(src + i8 + 4);
  uint4 p;
  p.x = pack2(f0.x, f0.y); p.y = pack2(f0.z, f0.w);
  p.z = pack2(f1.x, f1.y); p.w = pack2(f1.z, f1.w);
  *(uint4*)(dst + i8) = p;
}

// =========================================================================
// Stage 0b: w1 fp32 [512][1024][3][3] -> bf16 k'-ordered [512][rs*1024+ci]
// =========================================================================
__global__ __launch_bounds__(256)
void prep_w1_bf16(const float* __restrict__ w1, __bf16* __restrict__ w1r) {
  const int m   = blockIdx.x;
  const int tid = threadIdx.x;
#pragma unroll
  for (int i = 0; i < 4; ++i) {
    int ci = tid + 256 * i;
    const float* src = w1 + (size_t)m * K1 + ci * 9;
    __bf16* dstm = w1r + (size_t)m * K1 + ci;
#pragma unroll
    for (int rs = 0; rs < 9; ++rs)
      dstm[rs * 1024] = f2bf(src[rs]);
  }
}

// =========================================================================
// Stage 1 (fast path): implicit-GEMM conv3x3 + bias + ReLU6, rs-major K order.
// Explicit LDS double buffering, one workgroup barrier per K-step.
// A tile (L2-hot prefetched stream) is committed to buf[nxt] immediately at
// load time; only the 8-VGPR packed B stage crosses the WMMA block (no spill).
// =========================================================================
__global__ __launch_bounds__(256)
void rpn_conv3x3_fast(const unsigned short* __restrict__ fmapT,  // bf16 bits, NCHW
                      const __bf16* __restrict__ w1r,            // bf16, k'-ordered
                      const float* __restrict__ b1,
                      __bf16* __restrict__ xout) {
  __shared__ __bf16 As[2][BM][LDK];
  __shared__ __bf16 Bs[2][BN][LDK];

  const int tid    = threadIdx.x;
  const int tile_n = blockIdx.x * BN;
  const int tile_m = blockIdx.y * BM;

  // ---- B-loader decomposition: lanes sweep n (coalesced), tid>>5 picks ci sub-block
  const int nlane = tid & 31;
  const int cb4   = (tid >> 5) * 4;          // ci offset within tile: 0,4,...,28
  int bb[4], hh[4], ww[4], nvalid = 0;
#pragma unroll
  for (int jn = 0; jn < 4; ++jn) {
    int row = nlane + 32 * jn;               // 0..127
    int n   = tile_n + row;
    int v   = (n < NTOT) ? 1 : 0;
    int nc  = v ? n : 0;
    int b   = nc / HW;
    int rm  = nc - b * HW;
    int h   = rm / W_;
    int w   = rm - h * W_;
    bb[jn] = b * (CIN * HW);
    hh[jn] = h; ww[jn] = w;
    nvalid |= v << jn;
  }

  // ---- A-loader fixed per-thread offsets (2 uint4 chunks of the 128x32 tile)
  const __bf16* w1rp = w1r + (size_t)tile_m * K1;
  int arow[2], aq[2];
  size_t aoffth[2];
#pragma unroll
  for (int i = 0; i < 2; ++i) {
    int c   = tid + 256 * i;
    arow[i] = c >> 2;
    aq[i]   = (c & 3) * 8;
    aoffth[i] = (size_t)arow[i] * K1 + aq[i];
  }

  const int lane = tid & 31;
  const int wid  = tid >> 5;
  const int wm   = wid & 3;
  const int wn   = wid >> 2;
  const int l16  = lane & 15;
  const int hi   = lane >> 4;
  const int kb   = hi * 8;

  v8f acc[2][4];
#pragma unroll
  for (int i = 0; i < 2; ++i)
#pragma unroll
    for (int j = 0; j < 4; ++j) {
      v8f z = {0.f, 0.f, 0.f, 0.f, 0.f, 0.f, 0.f, 0.f};
      acc[i][j] = z;
    }

  // ---- tap state for rs=0 (r=0, s=0)
  int voff[4], okm = 0;
#pragma unroll
  for (int jn = 0; jn < 4; ++jn) {
    int y = hh[jn] - 1, x = ww[jn] - 1;
    bool ok = ((nvalid >> jn) & 1) && ((unsigned)y < (unsigned)H_) &&
              ((unsigned)x < (unsigned)W_);
    voff[jn] = bb[jn] + cb4 * HW + y * W_ + x;
    okm |= (int)ok << jn;
  }

  // ---- prologue: stage step 0 into buffer 0
  uint2 bR[4];
  {
    uint4 a0 = *(const uint4*)(w1rp + aoffth[0]);
    uint4 a1 = *(const uint4*)(w1rp + aoffth[1]);
    *(uint4*)&As[0][arow[0]][aq[0]] = a0;
    *(uint4*)&As[0][arow[1]][aq[1]] = a1;
#pragma unroll
    for (int jn = 0; jn < 4; ++jn) {
      unsigned lo = 0u, hi2 = 0u;
      if ((okm >> jn) & 1) {
        const unsigned short* p = fmapT + voff[jn];
        lo  = (unsigned)p[0]      | ((unsigned)p[HW]     << 16);
        hi2 = (unsigned)p[2 * HW] | ((unsigned)p[3 * HW] << 16);
      }
      bR[jn].x = lo; bR[jn].y = hi2;
      *(uint2*)&Bs[0][nlane + 32 * jn][cb4] = bR[jn];
    }
  }
  __syncthreads();

  int rs = 0, ci0 = 0;
#pragma unroll 2
  for (int t = 0; t < NSTEPS; ++t) {
    const int  cur  = t & 1;
    const int  nxt  = cur ^ 1;
    const bool more = (t + 1) < NSTEPS;

    if (more) {
      ci0 += BK;
      if (ci0 == CIN) {                    // uniform branch, taken 8 times total
        ci0 = 0;
        ++rs;
        const int r = rs / 3, s = rs - 3 * r;
        int okn = 0;
#pragma unroll
        for (int jn = 0; jn < 4; ++jn) {
          int y = hh[jn] + r - 1, x = ww[jn] + s - 1;
          bool ok = ((nvalid >> jn) & 1) && ((unsigned)y < (unsigned)H_) &&
                    ((unsigned)x < (unsigned)W_);
          voff[jn] = bb[jn] + cb4 * HW + y * W_ + x;
          okn |= (int)ok << jn;
        }
        okm = okn;
      }
      // ---- A: load and commit straight to buf[nxt] (L2-hot, prefetched stream;
      //      buf[nxt] was last read before the previous barrier -> safe)
      {
        const __bf16* ap = w1rp + rs * 1024 + ci0;
        uint4 a0 = *(const uint4*)(ap + aoffth[0]);
        uint4 a1 = *(const uint4*)(ap + aoffth[1]);
        *(uint4*)&As[nxt][arow[0]][aq[0]] = a0;
        *(uint4*)&As[nxt][arow[1]][aq[1]] = a1;
      }
      // ---- B: issue gather + pack into 8 VGPRs (committed after the WMMAs)
      {
        const unsigned short* fs = fmapT + (size_t)ci0 * HW;
#pragma unroll
        for (int jn = 0; jn < 4; ++jn) {
          unsigned lo = 0u, hi2 = 0u;
          if ((okm >> jn) & 1) {
            const unsigned short* p = fs + voff[jn];
            lo  = (unsigned)p[0]      | ((unsigned)p[HW]     << 16);
            hi2 = (unsigned)p[2 * HW] | ((unsigned)p[3 * HW] << 16);
          }
          bR[jn].x = lo; bR[jn].y = hi2;
        }
      }
      if (ci0 + BK < CIN)                  // L2 prefetch of the pure A stream
        __builtin_prefetch(w1rp + rs * 1024 + ci0 + BK + aoffth[0], 0, 1);
    }
    __builtin_amdgcn_sched_barrier(0);     // staging may not sink below here

    // ---- compute current tile from buf[cur]; one B fragment live at a time
    {
      const int row0 = wm * 32 + l16;
      Frag a0, a1;
      a0.h[0] = *(const v8bf*)&As[cur][row0][kb];
      a0.h[1] = *(const v8bf*)&As[cur][row0][16 + kb];
      a1.h[0] = *(const v8bf*)&As[cur][row0 + 16][kb];
      a1.h[1] = *(const v8bf*)&As[cur][row0 + 16][16 + kb];
#pragma unroll
      for (int j = 0; j < 4; ++j) {
        Frag bq;
        int col = wn * 64 + j * 16 + l16;
        bq.h[0] = *(const v8bf*)&Bs[cur][col][kb];
        bq.h[1] = *(const v8bf*)&Bs[cur][col][16 + kb];
        acc[0][j] = __builtin_amdgcn_wmma_f32_16x16x32_bf16(
            false, a0.v, false, bq.v, (short)0, acc[0][j], false, false);
        acc[1][j] = __builtin_amdgcn_wmma_f32_16x16x32_bf16(
            false, a1.v, false, bq.v, (short)0, acc[1][j], false, false);
      }
    }
    __builtin_amdgcn_sched_barrier(0);     // B commit stays below the WMMAs

    // ---- commit packed B stage to buf[nxt]; one barrier per step
    if (more) {
#pragma unroll
      for (int jn = 0; jn < 4; ++jn)
        *(uint2*)&Bs[nxt][nlane + 32 * jn][cb4] = bR[jn];
    }
    __syncthreads();
  }

  // ---- epilogue: bias + ReLU6 -> bf16 NHWC, one 16B store per fragment/lane
#pragma unroll
  for (int i = 0; i < 2; ++i) {
    int mb = tile_m + wm * 32 + i * 16 + 8 * hi;
    float4 c0 = *(const float4*)(b1 + mb);
    float4 c1 = *(const float4*)(b1 + mb + 4);
    float bias[8] = {c0.x, c0.y, c0.z, c0.w, c1.x, c1.y, c1.z, c1.w};
#pragma unroll
    for (int j = 0; j < 4; ++j) {
      int n = tile_n + wn * 64 + j * 16 + l16;
      if (n < NTOT) {
        Pack8 p;
#pragma unroll
        for (int rr = 0; rr < 8; ++rr) {
          float v = acc[i][j][rr] + bias[rr];
          v = fminf(fmaxf(v, 0.0f), 6.0f);
          p.b[rr] = f2bf(v);
        }
        *(uint4*)(xout + (size_t)n * DIM + mb) = p.u;
      }
    }
  }
}

// =========================================================================
// Stage 1 (fallback, small d_ws): direct im2col kernel (fp32 inputs)
// =========================================================================
__global__ __launch_bounds__(256)
void rpn_conv3x3_relu6(const float* __restrict__ fmap,
                       const float* __restrict__ w1,
                       const float* __restrict__ b1,
                       __bf16* __restrict__ xout) {
  __shared__ __bf16 As[BM][LDK];
  __shared__ __bf16 Bs[BN][LDK];

  const int tid    = threadIdx.x;
  const int tile_n = blockIdx.x * BN;
  const int tile_m = blockIdx.y * BM;

  const int kk    = tid & 31;
  const int nrow0 = tid >> 5;
  int gbase[16], hv[16], wv[16], vmask = 0;
#pragma unroll
  for (int j = 0; j < 16; ++j) {
    int nl = nrow0 + 8 * j;
    int n  = tile_n + nl;
    int v  = (n < NTOT) ? 1 : 0;
    int nc = v ? n : 0;
    int b  = nc / HW;
    int rm = nc - b * HW;
    int h  = rm / W_;
    int w  = rm - h * W_;
    gbase[j] = b * (CIN * HW) + h * W_ + w;
    hv[j] = h; wv[j] = w;
    vmask |= v << j;
  }

  const int lane = tid & 31;
  const int wid  = tid >> 5;
  const int wm   = wid & 3;
  const int wn   = wid >> 2;
  const int l16  = lane & 15;
  const int hi   = lane >> 4;
  const int kb   = hi * 8;

  v8f acc[2][4];
#pragma unroll
  for (int i = 0; i < 2; ++i)
#pragma unroll
    for (int j = 0; j < 4; ++j) {
      v8f z = {0.f, 0.f, 0.f, 0.f, 0.f, 0.f, 0.f, 0.f};
      acc[i][j] = z;
    }

  for (int k0 = 0; k0 < K1; k0 += BK) {
#pragma unroll
    for (int i = 0; i < 4; ++i) {
      int c   = tid + 256 * i;
      int row = c >> 3;
      int kq  = (c & 7) * 4;
      const float4 f = *(const float4*)(w1 + (size_t)(tile_m + row) * K1 + k0 + kq);
      uint2 p; p.x = pack2(f.x, f.y); p.y = pack2(f.z, f.w);
      *(uint2*)&As[row][kq] = p;
    }
    {
      int kg = k0 + kk;
      int ci = kg / 9;
      int rsx = kg - ci * 9;
      int r  = rsx / 3;
      int s  = rsx - r * 3;
      int koff = ci * HW + (r - 1) * W_ + (s - 1);
#pragma unroll
      for (int j = 0; j < 16; ++j) {
        int  y  = hv[j] + r - 1;
        int  x  = wv[j] + s - 1;
        bool ok = ((vmask >> j) & 1) && ((unsigned)y < (unsigned)H_) &&
                  ((unsigned)x < (unsigned)W_);
        float val = ok ? fmap[gbase[j] + koff] : 0.0f;
        Bs[nrow0 + 8 * j][kk] = f2bf(val);
      }
    }
    __syncthreads();

    Frag a[2], bq[4];
#pragma unroll
    for (int i = 0; i < 2; ++i) {
      int row = wm * 32 + i * 16 + l16;
      a[i].h[0] = *(const v8bf*)&As[row][kb];
      a[i].h[1] = *(const v8bf*)&As[row][16 + kb];
    }
#pragma unroll
    for (int j = 0; j < 4; ++j) {
      int col = wn * 64 + j * 16 + l16;
      bq[j].h[0] = *(const v8bf*)&Bs[col][kb];
      bq[j].h[1] = *(const v8bf*)&Bs[col][16 + kb];
    }
#pragma unroll
    for (int i = 0; i < 2; ++i)
#pragma unroll
      for (int j = 0; j < 4; ++j)
        acc[i][j] = __builtin_amdgcn_wmma_f32_16x16x32_bf16(
            false, a[i].v, false, bq[j].v, (short)0, acc[i][j], false, false);
    __syncthreads();
  }

#pragma unroll
  for (int i = 0; i < 2; ++i) {
    int mb = tile_m + wm * 32 + i * 16 + 8 * hi;
    float4 c0 = *(const float4*)(b1 + mb);
    float4 c1 = *(const float4*)(b1 + mb + 4);
    float bias[8] = {c0.x, c0.y, c0.z, c0.w, c1.x, c1.y, c1.z, c1.w};
#pragma unroll
    for (int j = 0; j < 4; ++j) {
      int n = tile_n + wn * 64 + j * 16 + l16;
      if (n < NTOT) {
        Pack8 p;
#pragma unroll
        for (int rr = 0; rr < 8; ++rr) {
          float v = acc[i][j][rr] + bias[rr];
          v = fminf(fmaxf(v, 0.0f), 6.0f);
          p.b[rr] = f2bf(v);
        }
        *(uint4*)(xout + (size_t)n * DIM + mb) = p.u;
      }
    }
  }
}

// =========================================================================
// Stage 2: conv1x1 GEMM  M=120(pad 128), K=512, N=43808  + bias -> fp32 NHWC
// =========================================================================
__global__ __launch_bounds__(256)
void rpn_conv1x1_bias(const __bf16* __restrict__ xin,
                      const float* __restrict__ w2,
                      const float* __restrict__ b2,
                      float* __restrict__ out) {
  __shared__ __bf16 As[BM][LDK];
  __shared__ __bf16 Bs[BN][LDK];

  const int tid    = threadIdx.x;
  const int tile_n = blockIdx.x * BN;
  const int lane   = tid & 31;
  const int wid    = tid >> 5;
  const int wm     = wid & 3;
  const int wn     = wid >> 2;
  const int l16    = lane & 15;
  const int hi     = lane >> 4;
  const int kb     = hi * 8;

  v8f acc[2][4];
#pragma unroll
  for (int i = 0; i < 2; ++i)
#pragma unroll
    for (int j = 0; j < 4; ++j) {
      v8f z = {0.f, 0.f, 0.f, 0.f, 0.f, 0.f, 0.f, 0.f};
      acc[i][j] = z;
    }

  for (int k0 = 0; k0 < DIM; k0 += BK) {
#pragma unroll
    for (int i = 0; i < 4; ++i) {
      int c   = tid + 256 * i;
      int row = c >> 3;
      int kq  = (c & 7) * 4;
      float4 f = {0.f, 0.f, 0.f, 0.f};
      if (row < COUT) f = *(const float4*)(w2 + (size_t)row * DIM + k0 + kq);
      uint2 p; p.x = pack2(f.x, f.y); p.y = pack2(f.z, f.w);
      *(uint2*)&As[row][kq] = p;
    }
#pragma unroll
    for (int i = 0; i < 2; ++i) {
      int c  = tid + 256 * i;
      int nl = c >> 2;
      int q  = (c & 3) * 8;
      int n  = tile_n + nl;
      uint4 d = {0u, 0u, 0u, 0u};
      if (n < NTOT) d = *(const uint4*)(xin + (size_t)n * DIM + k0 + q);
      *(uint4*)&Bs[nl][q] = d;
    }
    __syncthreads();

    Frag a[2], bq[4];
#pragma unroll
    for (int i = 0; i < 2; ++i) {
      int row = wm * 32 + i * 16 + l16;
      a[i].h[0] = *(const v8bf*)&As[row][kb];
      a[i].h[1] = *(const v8bf*)&As[row][16 + kb];
    }
#pragma unroll
    for (int j = 0; j < 4; ++j) {
      int col = wn * 64 + j * 16 + l16;
      bq[j].h[0] = *(const v8bf*)&Bs[col][kb];
      bq[j].h[1] = *(const v8bf*)&Bs[col][16 + kb];
    }
#pragma unroll
    for (int i = 0; i < 2; ++i)
#pragma unroll
      for (int j = 0; j < 4; ++j)
        acc[i][j] = __builtin_amdgcn_wmma_f32_16x16x32_bf16(
            false, a[i].v, false, bq[j].v, (short)0, acc[i][j], false, false);
    __syncthreads();
  }

#pragma unroll
  for (int i = 0; i < 2; ++i) {
    int mb = wm * 32 + i * 16 + 8 * hi;
    if (mb < COUT) {
      float4 c0 = *(const float4*)(b2 + mb);
      float4 c1 = *(const float4*)(b2 + mb + 4);
#pragma unroll
      for (int j = 0; j < 4; ++j) {
        int n = tile_n + wn * 64 + j * 16 + l16;
        if (n < NTOT) {
          float* dst = out + (size_t)n * COUT + mb;
          float4 o0, o1;
          o0.x = acc[i][j][0] + c0.x; o0.y = acc[i][j][1] + c0.y;
          o0.z = acc[i][j][2] + c0.z; o0.w = acc[i][j][3] + c0.w;
          o1.x = acc[i][j][4] + c1.x; o1.y = acc[i][j][5] + c1.y;
          o1.z = acc[i][j][6] + c1.z; o1.w = acc[i][j][7] + c1.w;
          *(float4*)dst       = o0;
          *(float4*)(dst + 4) = o1;
        }
      }
    }
  }
}

extern "C" void kernel_launch(void* const* d_in, const int* in_sizes, int n_in,
                              void* d_out, int out_size, void* d_ws, size_t ws_size,
                              hipStream_t stream) {
  const float* fmap = (const float*)d_in[0];
  const float* w1   = (const float*)d_in[1];
  const float* b1   = (const float*)d_in[2];
  const float* w2   = (const float*)d_in[3];
  const float* b2   = (const float*)d_in[4];
  float* out = (float*)d_out;

  const size_t XINT_BYTES = (size_t)NTOT * DIM * 2;          // 44,859,392
  const size_t FMT_BYTES  = (size_t)B_ * CIN * HW * 2;       // 89,718,784
  const size_t W1R_BYTES  = (size_t)DIM * K1 * 2;            //  9,437,184

  char*   wsb  = (char*)d_ws;
  __bf16* xint = (__bf16*)wsb;                               // [NTOT][DIM] bf16 NHWC

  const dim3 g1((NTOT + BN - 1) / BN, DIM / BM);             // (343, 4)
  const dim3 g2((NTOT + BN - 1) / BN, 1);                    // (343)

  if (ws_size >= XINT_BYTES + FMT_BYTES + W1R_BYTES) {
    unsigned short* fmapT = (unsigned short*)(wsb + XINT_BYTES);
    __bf16*         w1r   = (__bf16*)(wsb + XINT_BYTES + FMT_BYTES);
    prep_fmap_bf16<<<(unsigned)((size_t)B_ * CIN * HW / (256 * 8)), 256, 0, stream>>>(fmap, fmapT);
    prep_w1_bf16<<<DIM, 256, 0, stream>>>(w1, w1r);
    rpn_conv3x3_fast<<<g1, 256, 0, stream>>>(fmapT, w1r, b1, xint);
  } else {
    rpn_conv3x3_relu6<<<g1, 256, 0, stream>>>(fmap, w1, b1, xint);
  }
  rpn_conv1x1_bias<<<g2, 256, 0, stream>>>(xint, w2, b2, out);
}
